// TilinGNN_9225589752440
// MI455X (gfx1250) — compile-verified, matrix operand order checked by reading, not run on hardware
//
#include <hip/hip_runtime.h>
#include <math.h>

// ---------------------------------------------------------------------------
// TilinGNN forward on gfx1250 (MI455X). All GEMMs via v_wmma_f32_16x16x32_f16
// with 2x2 register tiling per wave (4 WMMAs per 8 b128 operand loads).
// B (weights) pre-packed into per-lane WMMA operand layout -> contiguous b128
// loads; A fragments loaded as 2x b128. BN-apply fuses leaky-ReLU and (for
// edge MLPs) the atomic scatter of the segment-sum.
// ---------------------------------------------------------------------------

typedef _Float16 h16;
typedef __attribute__((ext_vector_type(16))) _Float16 v16h;
typedef __attribute__((ext_vector_type(8)))  _Float16 h8v;
typedef __attribute__((ext_vector_type(4)))  _Float16 h4v;
typedef __attribute__((ext_vector_type(8)))  float    v8f;

#define GNN_N   50000
#define GNN_NF  1025
#define GNN_NFP 1056   // 1025 padded to mult of 32
#define GNN_W   128
#define GNN_EF  8
#define GNN_D   8
#define GNN_EA  400000
#define GNN_EC  800000
#define GNN_KE  288    // 2*W+EF = 264 padded to mult of 32
#define GNN_KN  256
#define GNN_FIN 1152   // (D+1)*W
#define GNN_SLOPE 0.01f
#define GNN_EPS   1e-5f

// ---------------------------------------------------------------------------
__global__ void k_zero(float* __restrict__ p, long n) {
  long i = (long)blockIdx.x * blockDim.x + threadIdx.x;
  if (i < n) p[i] = 0.0f;
}

// f32 -> f16 with zero padding (row-major both sides)
__global__ void k_cvt_pad(const float* __restrict__ in, h16* __restrict__ out,
                          int rows_out, int cols_out, int rows_in, int cols_in) {
  long i = (long)blockIdx.x * blockDim.x + threadIdx.x;
  long tot = (long)rows_out * cols_out;
  if (i >= tot) return;
  int c = (int)(i % cols_out);
  long r = i / cols_out;
  float v = 0.0f;
  if (r < rows_in && c < cols_in) v = in[r * (long)cols_in + c];
  out[i] = (h16)v;
}

__global__ void k_copy_strided(const float* __restrict__ src, int sstride,
                               float* __restrict__ dst, int dstride, long rows, int cols) {
  long i = (long)blockIdx.x * blockDim.x + threadIdx.x;
  if (i >= rows * (long)cols) return;
  int c = (int)(i % cols);
  long r = i / cols;
  dst[r * (long)dstride + c] = src[r * (long)sstride + c];
}

// ---------------------------------------------------------------------------
// Pack f32 weight [K,Nc] (zero-padded to Kpad rows) into WMMA B-operand
// fragments: out index = (((nt*KT + kt)*32 + lane)*16 + i), where the lane's
// fragment covers n = nt*16 + (lane&15), k = kt*32 + (lane>>4)*8 + (i<8?i:i+8)
// (CDNA5 ISA 16-bit operand layout, CDNA5 ISA 7.12.2).
__global__ void k_pack_B(const float* __restrict__ w, h16* __restrict__ out,
                         int K, int Kpad, int Nc) {
  long idx = (long)blockIdx.x * blockDim.x + threadIdx.x;
  long tot = (long)Kpad * Nc;
  if (idx >= tot) return;
  int i    = (int)(idx & 15);
  int lane = (int)((idx >> 4) & 31);
  long t   = idx >> 9;                // nt*KT + kt
  int KT   = Kpad >> 5;
  int kt   = (int)(t % KT);
  int nt   = (int)(t / KT);
  int n = nt * 16 + (lane & 15);
  int k = kt * 32 + ((lane >> 4) << 3) + (i < 8 ? i : i + 8);
  float v = (k < K) ? w[(long)k * Nc + n] : 0.0f;
  out[idx] = (h16)v;
}

// ---------------------------------------------------------------------------
// WMMA GEMM with 2x2 register tiling: each wave computes a 32x32 output tile
// (2 M-tiles x 2 N-tiles), reusing each A/B fragment across two WMMAs.
// 8 waves / block cover a 256-row slab of one 32-column slab.
// A row reads clamp (keeps EXEC all-ones for WMMA); stores guard m < M.
__global__ __launch_bounds__(256) void k_wmma_gemm(
    const h16* __restrict__ A, const h16* __restrict__ Bpack,
    const float* __restrict__ bias, float* __restrict__ C,
    int M, int K, int Nc, int lrelu) {
  int lane = threadIdx.x & 31;
  int wave = threadIdx.x >> 5;
  long slab = (long)blockIdx.x * 8 + wave;      // 32-row slab index
  if (slab * 32 >= M) return;                   // wave-uniform
  int l15 = lane & 15, hi = lane >> 4;
  int KT = K >> 5;
  int nt0 = blockIdx.y * 2;

  long m0 = slab * 32;
  long r0 = m0 + l15;       if (r0 >= M) r0 = M - 1;
  long r1 = m0 + 16 + l15;  if (r1 >= M) r1 = M - 1;
  const h16*  A0 = A + r0 * K + hi * 8;
  const h16*  A1 = A + r1 * K + hi * 8;
  const v16h* B0 = (const v16h*)Bpack + (long)nt0 * KT * 32 + lane;
  const v16h* B1 = B0 + (long)KT * 32;

  v8f acc[2][2] = {};
  for (int kt = 0; kt < KT; ++kt) {
    union { v16h v; h8v h[2]; } a0, a1;
    a0.h[0] = *(const h8v*)(A0 + (long)kt * 32);       // K = kt*32 + hi*8 + 0..7
    a0.h[1] = *(const h8v*)(A0 + (long)kt * 32 + 16);  // K = kt*32 + 16 + hi*8 + 0..7
    a1.h[0] = *(const h8v*)(A1 + (long)kt * 32);
    a1.h[1] = *(const h8v*)(A1 + (long)kt * 32 + 16);
    v16h b0 = B0[(long)kt * 32];
    v16h b1 = B1[(long)kt * 32];
    acc[0][0] = __builtin_amdgcn_wmma_f32_16x16x32_f16(
        false, a0.v, false, b0, (short)0, acc[0][0], false, false);
    acc[0][1] = __builtin_amdgcn_wmma_f32_16x16x32_f16(
        false, a0.v, false, b1, (short)0, acc[0][1], false, false);
    acc[1][0] = __builtin_amdgcn_wmma_f32_16x16x32_f16(
        false, a1.v, false, b0, (short)0, acc[1][0], false, false);
    acc[1][1] = __builtin_amdgcn_wmma_f32_16x16x32_f16(
        false, a1.v, false, b1, (short)0, acc[1][1], false, false);
  }

#pragma unroll
  for (int tj = 0; tj < 2; ++tj) {
    float bc = bias ? bias[(nt0 + tj) * 16 + l15] : 0.0f;
    long cc = (long)(nt0 + tj) * 16 + l15;
#pragma unroll
    for (int ti = 0; ti < 2; ++ti) {
#pragma unroll
      for (int r = 0; r < 8; ++r) {
        long m = m0 + ti * 16 + r + hi * 8;   // C/D layout: VGPR r -> M = r + 8*hi
        if (m < M) {
          float v = acc[ti][tj][r] + bc;
          if (lrelu) v = v > 0.0f ? v : GNN_SLOPE * v;
          C[m * (long)Nc + cc] = v;
        }
      }
    }
  }
}

// ---------------------------------------------------------------------------
// Column sums / sums-of-squares for BatchNorm (Nc power of two <= 256).
__global__ __launch_bounds__(256) void k_col_stats(
    const float* __restrict__ C, long M, int Nc, float* __restrict__ stats) {
  __shared__ float ls[512];
  int tid  = threadIdx.x;
  int col  = tid & (Nc - 1);
  int rsub = tid / Nc;
  int rstep = 256 / Nc;
  long base = (long)blockIdx.x * 256;
  float s = 0.0f, s2 = 0.0f;
  for (int r = rsub; r < 256; r += rstep) {
    long rr = base + r;
    if (rr < M) {
      float v = C[rr * (long)Nc + col];
      s += v; s2 += v * v;
    }
  }
  ls[tid] = s; ls[256 + tid] = s2;
  __syncthreads();
  for (int off = 128; off >= Nc; off >>= 1) {
    if (tid < off) { ls[tid] += ls[tid + off]; ls[256 + tid] += ls[256 + tid + off]; }
    __syncthreads();
  }
  if (tid < Nc) {
    atomicAdd(&stats[tid], ls[tid]);
    atomicAdd(&stats[Nc + tid], ls[256 + tid]);
  }
}

// BN apply + leaky-ReLU, 4 columns per thread; optional strided f32 out,
// f16 out, and atomic scatter into segment sums.
__global__ void k_bn4(const float* __restrict__ P, long M, int Nc,
                      const float* __restrict__ stats,
                      const float* __restrict__ g, const float* __restrict__ be,
                      float* __restrict__ o32, int o32s,
                      h16* __restrict__ o16,
                      const int* __restrict__ sdst, float* __restrict__ ssum) {
  int Nc4 = Nc >> 2;
  long i = (long)blockIdx.x * blockDim.x + threadIdx.x;
  if (i >= M * (long)Nc4) return;
  int c = (int)(i % Nc4) << 2;
  long r = i / Nc4;
  float invM = 1.0f / (float)M;
  int d = sdst ? sdst[r] : 0;
  float4 pv = *(const float4*)(P + r * (long)Nc + c);
  float vv[4] = {pv.x, pv.y, pv.z, pv.w};
  h4v o16v;
#pragma unroll
  for (int j = 0; j < 4; ++j) {
    int cc = c + j;
    float mean = stats[cc] * invM;
    float var  = fmaxf(stats[Nc + cc] * invM - mean * mean, 0.0f);
    float v = (vv[j] - mean) * rsqrtf(var + GNN_EPS) * g[cc] + be[cc];
    v = v > 0.0f ? v : GNN_SLOPE * v;
    vv[j] = v;
    o16v[j] = (h16)v;
    if (sdst) atomicAdd(&ssum[(long)d * Nc + cc], v);
  }
  if (o32) {
    float4 ov = {vv[0], vv[1], vv[2], vv[3]};
    *(float4*)(o32 + r * (long)o32s + c) = ov;
  }
  if (o16) *(h4v*)(o16 + r * (long)Nc + c) = o16v;
}

// ---------------------------------------------------------------------------
// Edge-input gather, 4 halves per thread: [h[src] | h[dst] | efeat | 0-pad]
__global__ void k_gather_edge4(const float* __restrict__ h, int hstride,
                               const float* __restrict__ ef,
                               const int* __restrict__ src, const int* __restrict__ dst,
                               long E, int Kpad, h16* __restrict__ out) {
  int Kp4 = Kpad >> 2;
  long i = (long)blockIdx.x * blockDim.x + threadIdx.x;
  if (i >= E * (long)Kp4) return;
  int c = (int)(i % Kp4) << 2;
  long e = i / Kp4;
  float4 v = {0.f, 0.f, 0.f, 0.f};
  if (c < GNN_W)            v = *(const float4*)(h + (long)src[e] * hstride + c);
  else if (c < 2 * GNN_W)   v = *(const float4*)(h + (long)dst[e] * hstride + (c - GNN_W));
  else if (ef && c < 2 * GNN_W + GNN_EF)
                            v = *(const float4*)(ef + e * (long)GNN_EF + (c - 2 * GNN_W));
  h4v o = {(h16)v.x, (h16)v.y, (h16)v.z, (h16)v.w};
  *(h4v*)(out + e * (long)Kpad + c) = o;
}

// Node-input gather, 4 halves per thread: [h | segment-mean(agg)]
__global__ void k_gather_node4(const float* __restrict__ h, int hstride,
                               const float* __restrict__ aggSum,
                               const float* __restrict__ cnt,
                               long n, h16* __restrict__ out) {
  long i = (long)blockIdx.x * blockDim.x + threadIdx.x;
  if (i >= n * 64L) return;                 // 256/4 groups per row
  int c = (int)(i & 63) << 2;
  long r = i >> 6;
  float4 v;
  if (c < GNN_W) v = *(const float4*)(h + r * (long)hstride + c);
  else {
    float ct = cnt[r]; ct = ct < 1.0f ? 1.0f : ct;
    float4 s = *(const float4*)(aggSum + r * (long)GNN_W + (c - GNN_W));
    v.x = s.x / ct; v.y = s.y / ct; v.z = s.z / ct; v.w = s.w / ct;
  }
  h4v o = {(h16)v.x, (h16)v.y, (h16)v.z, (h16)v.w};
  *(h4v*)(out + r * 256L + c) = o;
}

__global__ void k_count(const int* __restrict__ dst, long E, float* __restrict__ cnt) {
  long i = (long)blockIdx.x * blockDim.x + threadIdx.x;
  if (i < E) atomicAdd(&cnt[dst[i]], 1.0f);
}

// Gating + residual skip; writes new b1 into mids slice (layer+1), updates b2
__global__ void k_gate(const float* __restrict__ gout, const float* __restrict__ cout,
                       float* __restrict__ mids, float* __restrict__ b2,
                       long n, int layer) {
  long i = (long)blockIdx.x * blockDim.x + threadIdx.x;
  if (i >= n * (long)GNN_W) return;
  int c = (int)(i & (GNN_W - 1));
  long r = i >> 7;
  float gv = gout[i], cv = cout[i];
  float v = gv * cv;
  if (layer >= 2) v += mids[r * (long)GNN_FIN + (long)(layer - 2) * GNN_W + c];
  mids[r * (long)GNN_FIN + (long)(layer + 1) * GNN_W + c] = v;
  b2[i] = cv;
}

// Final projection [128]->1 + sigmoid
__global__ __launch_bounds__(128) void k_out(const float* __restrict__ h,
                                             const float* __restrict__ w,
                                             const float* __restrict__ b,
                                             float* __restrict__ out, long n) {
  __shared__ float ls[128];
  long nn = blockIdx.x;
  int t = threadIdx.x;
  ls[t] = h[nn * (long)GNN_W + t] * w[t];
  __syncthreads();
  for (int off = 64; off > 0; off >>= 1) {
    if (t < off) ls[t] += ls[t + off];
    __syncthreads();
  }
  if (t == 0) out[nn] = 1.0f / (1.0f + expf(-(ls[0] + b[0])));
}

// ---------------------------------------------------------------------------
extern "C" void kernel_launch(void* const* d_in, const int* in_sizes, int n_in,
                              void* d_out, int out_size, void* d_ws, size_t ws_size,
                              hipStream_t stream) {
  (void)in_sizes; (void)n_in; (void)out_size; (void)ws_size;

  const float* x      = (const float*)d_in[0];
  const int*   adjIdx = (const int*)d_in[1];
  const float* adjFeat= (const float*)d_in[2];
  const int*   colIdx = (const int*)d_in[3];
  const int* aSrc = adjIdx, * aDst = adjIdx + GNN_EA;
  const int* cSrc = colIdx, * cDst = colIdx + GNN_EC;

  const int PI_INIT = 4, PI_GCONV = 12, PI_CCONV = 140, PI_FINAL = 268, PI_OUT = 276;
  auto F = [&](int idx) { return (const float*)d_in[idx]; };

  // -------- workspace bump allocator (256B aligned) --------
  char* wsb = (char*)d_ws;
  size_t off = 0;
  auto alloc = [&](size_t bytes) -> void* {
    void* p = wsb + off;
    off += (bytes + 255) & ~(size_t)255;
    return p;
  };
  float* statsBuf = (float*)alloc(512 * sizeof(float));
  float* cnt      = (float*)alloc((size_t)GNN_N * sizeof(float));
  float* aggSum   = (float*)alloc((size_t)GNN_N * GNN_W * sizeof(float));
  float* b2       = (float*)alloc((size_t)GNN_N * GNN_W * sizeof(float));
  float* gout     = (float*)alloc((size_t)GNN_N * GNN_W * sizeof(float));
  float* cout     = (float*)alloc((size_t)GNN_N * GNN_W * sizeof(float));
  float* mids     = (float*)alloc((size_t)GNN_N * GNN_FIN * sizeof(float));
  h16*   A16      = (h16*)  alloc((size_t)GNN_EC * GNN_KN * sizeof(h16));
  float* bufP     = (float*)alloc((size_t)GNN_EC * GNN_W * sizeof(float));
  h16*   H16      = (h16*)  alloc((size_t)GNN_EC * GNN_W * sizeof(h16));

  auto gemm = [&](const h16* A, const h16* Bp, const float* bias, float* C,
                  long M, int K, int Nc, int lrelu) {
    long slabs = (M + 31) / 32;
    dim3 g((unsigned)((slabs + 7) / 8), (unsigned)(Nc / 32));
    k_wmma_gemm<<<g, 256, 0, stream>>>(A, Bp, bias, C, (int)M, K, Nc, lrelu);
  };
  auto zero = [&](float* p, long n) {
    k_zero<<<(unsigned)((n + 255) / 256), 256, 0, stream>>>(p, n);
  };
  auto stats = [&](const float* C, long M, int Nc) {
    zero(statsBuf, 2L * Nc);
    k_col_stats<<<(unsigned)((M + 255) / 256), 256, 0, stream>>>(C, M, Nc, statsBuf);
  };
  auto bn = [&](const float* P, long M, int Nc, const float* g, const float* be,
                float* o32, int o32s, h16* o16, const int* sdst, float* ssum) {
    long tot = M * (long)(Nc >> 2);
    k_bn4<<<(unsigned)((tot + 255) / 256), 256, 0, stream>>>(
        P, M, Nc, statsBuf, g, be, o32, o32s, o16, sdst, ssum);
  };
  auto cvt = [&](const float* in, h16* out, long ro, int co, long ri, int ci) {
    long tot = ro * (long)co;
    k_cvt_pad<<<(unsigned)((tot + 255) / 256), 256, 0, stream>>>(
        in, out, (int)ro, co, (int)ri, ci);
  };

  // -------- weight packing into WMMA B-operand layout --------
  auto packW = [&](int leaf, int K, int Kpad, int Nc) -> h16* {
    h16* p = (h16*)alloc((size_t)Kpad * Nc * sizeof(h16));
    long tot = (long)Kpad * Nc;
    k_pack_B<<<(unsigned)((tot + 255) / 256), 256, 0, stream>>>(
        F(leaf), p, K, Kpad, Nc);
    return p;
  };
  h16* wInit0 = packW(PI_INIT + 0, GNN_NF, GNN_NFP, GNN_W);
  h16* wInit1 = packW(PI_INIT + 4, GNN_W, GNN_W, GNN_W);
  h16 *wE0[GNN_D], *wE1[GNN_D], *wGN0[GNN_D], *wGN1[GNN_D];
  h16 *wM0[GNN_D], *wM1[GNN_D], *wCN0[GNN_D], *wCN1[GNN_D];
  for (int i = 0; i < GNN_D; ++i) {
    int gb = PI_GCONV + 16 * i;
    wE0[i]  = packW(gb + 0,  2 * GNN_W + GNN_EF, GNN_KE, GNN_W);
    wE1[i]  = packW(gb + 4,  GNN_W, GNN_W, GNN_W);
    wGN0[i] = packW(gb + 8,  GNN_KN, GNN_KN, GNN_W);
    wGN1[i] = packW(gb + 12, GNN_W, GNN_W, GNN_W);
    int cb = PI_CCONV + 16 * i;
    wM0[i]  = packW(cb + 0,  GNN_KN, GNN_KN, GNN_W);
    wM1[i]  = packW(cb + 4,  GNN_W, GNN_W, GNN_W);
    wCN0[i] = packW(cb + 8,  GNN_KN, GNN_KN, GNN_W);
    wCN1[i] = packW(cb + 12, GNN_W, GNN_W, GNN_W);
  }
  h16* wF0 = packW(PI_FINAL + 0, GNN_FIN, GNN_FIN, 256);
  h16* wF1 = packW(PI_FINAL + 2, 256, 256, 128);
  h16* wF2 = packW(PI_FINAL + 4, 128, 128, 64);
  h16* wF3 = packW(PI_FINAL + 6, 64, 64, 128);

  // 2-layer MLP with BN on both layers, input in A16 [M, Kpad]
  auto mlp2bn = [&](long M, int Kpad,
                    const h16* w0, int leaf0, const h16* w1, int leaf1,
                    float* o32, int o32s, const int* sdst, float* ssum) {
    gemm(A16, w0, F(leaf0 + 1), bufP, M, Kpad, GNN_W, 0);
    stats(bufP, M, GNN_W);
    bn(bufP, M, GNN_W, F(leaf0 + 2), F(leaf0 + 3), nullptr, 0, H16, nullptr, nullptr);
    gemm(H16, w1, F(leaf1 + 1), bufP, M, GNN_W, GNN_W, 0);
    stats(bufP, M, GNN_W);
    bn(bufP, M, GNN_W, F(leaf1 + 2), F(leaf1 + 3), o32, o32s, nullptr, sdst, ssum);
  };

  // -------- init MLP: x -> mids slice 0; b2 = same --------
  cvt(x, A16, GNN_N, GNN_NFP, GNN_N, GNN_NF);
  mlp2bn(GNN_N, GNN_NFP, wInit0, PI_INIT + 0, wInit1, PI_INIT + 4,
         mids, GNN_FIN, nullptr, nullptr);
  {
    long tot = (long)GNN_N * GNN_W;
    k_copy_strided<<<(unsigned)((tot + 255) / 256), 256, 0, stream>>>(
        mids, GNN_FIN, b2, GNN_W, GNN_N, GNN_W);
  }

  // -------- D message-passing layers --------
  for (int i = 0; i < GNN_D; ++i) {
    const float* b1 = mids + (long)i * GNN_W;   // row stride GNN_FIN
    int gb = PI_GCONV + 16 * i, cb = PI_CCONV + 16 * i;

    // --- graph conv on adjacency edges ---
    {
      long tot = (long)GNN_EA * (GNN_KE >> 2);
      k_gather_edge4<<<(unsigned)((tot + 255) / 256), 256, 0, stream>>>(
          b1, GNN_FIN, adjFeat, aSrc, aDst, GNN_EA, GNN_KE, A16);
    }
    zero(aggSum, (long)GNN_N * GNN_W);
    zero(cnt, GNN_N);
    k_count<<<(GNN_EA + 255) / 256, 256, 0, stream>>>(aDst, GNN_EA, cnt);
    mlp2bn(GNN_EA, GNN_KE, wE0[i], gb + 0, wE1[i], gb + 4,
           nullptr, 0, aDst, aggSum);
    {
      long tot = (long)GNN_N * 64;
      k_gather_node4<<<(unsigned)((tot + 255) / 256), 256, 0, stream>>>(
          b1, GNN_FIN, aggSum, cnt, GNN_N, A16);
    }
    mlp2bn(GNN_N, GNN_KN, wGN0[i], gb + 8, wGN1[i], gb + 12,
           gout, GNN_W, nullptr, nullptr);

    // --- collision conv on collision edges ---
    {
      long tot = (long)GNN_EC * (GNN_KN >> 2);
      k_gather_edge4<<<(unsigned)((tot + 255) / 256), 256, 0, stream>>>(
          b2, GNN_W, nullptr, cSrc, cDst, GNN_EC, GNN_KN, A16);
    }
    zero(aggSum, (long)GNN_N * GNN_W);
    zero(cnt, GNN_N);
    k_count<<<(GNN_EC + 255) / 256, 256, 0, stream>>>(cDst, GNN_EC, cnt);
    mlp2bn(GNN_EC, GNN_KN, wM0[i], cb + 0, wM1[i], cb + 4,
           nullptr, 0, cDst, aggSum);
    {
      long tot = (long)GNN_N * 64;
      k_gather_node4<<<(unsigned)((tot + 255) / 256), 256, 0, stream>>>(
          b2, GNN_W, aggSum, cnt, GNN_N, A16);
    }
    mlp2bn(GNN_N, GNN_KN, wCN0[i], cb + 8, wCN1[i], cb + 12,
           cout, GNN_W, nullptr, nullptr);

    // --- gate + residual -> mids slice i+1, update b2 ---
    {
      long tot = (long)GNN_N * GNN_W;
      k_gate<<<(unsigned)((tot + 255) / 256), 256, 0, stream>>>(
          gout, cout, mids, b2, GNN_N, i);
    }
  }

  // -------- final MLP (no BN, lrelu fused in GEMM epilogue) --------
  cvt(mids, A16, GNN_N, GNN_FIN, GNN_N, GNN_FIN);
  gemm(A16, wF0, F(PI_FINAL + 1), bufP, GNN_N, GNN_FIN, 256, 1);
  cvt(bufP, H16, GNN_N, 256, GNN_N, 256);
  gemm(H16, wF1, F(PI_FINAL + 3), bufP, GNN_N, 256, 128, 1);
  cvt(bufP, H16, GNN_N, 128, GNN_N, 128);
  gemm(H16, wF2, F(PI_FINAL + 5), bufP, GNN_N, 128, 64, 1);
  cvt(bufP, H16, GNN_N, 64, GNN_N, 64);
  gemm(H16, wF3, F(PI_FINAL + 7), bufP, GNN_N, 64, 128, 1);

  // -------- output head + passthrough of adj_e_features --------
  k_out<<<GNN_N, 128, 0, stream>>>(bufP, F(PI_OUT + 0), F(PI_OUT + 1),
                                   (float*)d_out, GNN_N);
  hipMemcpyAsync((char*)d_out + (size_t)GNN_N * sizeof(float), (const void*)adjFeat,
                 (size_t)GNN_EA * GNN_EF * sizeof(float),
                 hipMemcpyDeviceToDevice, stream);
}
